// CovGnn_2791728742623
// MI455X (gfx1250) — compile-verified, matrix operand order checked by reading, not run on hardware
//
#include <hip/hip_runtime.h>
#include <hip/hip_bf16.h>

// ---------------------------------------------------------------------------
// CovGnn (Chebyshev graph conv, order 4, 32 sampled neighbours) for gfx1250.
// N=6144, F_IN=F_OUT=256, ORDER=4, K_NEIGH=32.
// Heavy GEMMs: v_wmma_f32_16x16x32_bf16 with TDM (tensor_load_to_lds)
// double-buffered K=64 B-operand staging and software-pipelined A loads.
// ---------------------------------------------------------------------------

#define GN      6144
#define GF      256
#define GORDER  4
#define GK      32              // sampled neighbours per row
#define GKX     (GF * GORDER)   // 1024

#define KSTEP   64              // K elements staged per TDM transfer
// LDS row: 64 bf16 payload (128B) + 16B TDM pad = 144B pitch = 36 banks.
// gcd(36,64)=4 -> 16 consecutive rows hit distinct banks for ds_load_b128,
// and both K=32 subtile bases (byte 0 / 64) stay 16B aligned.
#define ROWP    72              // bf16 elements per padded LDS row

typedef __attribute__((ext_vector_type(16))) __bf16 v16bf;
typedef __attribute__((ext_vector_type(8)))  __bf16 v8bf;
typedef __attribute__((ext_vector_type(8)))  float  v8f;
typedef __attribute__((ext_vector_type(4)))  unsigned int u32x4;
typedef __attribute__((ext_vector_type(8)))  int    i32x8;
typedef __attribute__((ext_vector_type(4)))  int    i32x4;

#if defined(__has_builtin)
#if __has_builtin(__builtin_amdgcn_tensor_load_to_lds) && \
    __has_builtin(__builtin_amdgcn_s_wait_tensorcnt)
#define HAVE_TDM 1
#endif
#endif
#ifndef HAVE_TDM
#define HAVE_TDM 0
#endif

static __device__ __forceinline__ __bf16 f2bf(float f) { return (__bf16)f; }

// Load one 16-bit WMMA fragment from a row-major matrix (global or LDS).
// Caller passes pointer already offset to (row, k0 + (laneHi ? 8 : 0)).
// Fragment halves are 16 elements apart (K 0..7 & 16..23 per CDNA5 layout).
template <typename P>
static __device__ __forceinline__ v16bf ld_frag(const P* p) {
    v8bf lo = *(const v8bf*)(p);
    v8bf hi = *(const v8bf*)(p + 16);
    return __builtin_shufflevector(lo, hi,
        0, 1, 2, 3, 4, 5, 6, 7, 8, 9, 10, 11, 12, 13, 14, 15);
}

// ---------------------------------------------------------------------------
// TDM: async DMA of a 2-D bf16 tile [tile_d1 rows x tile_d0 elems] from a
// row-major [tensor_d1 x tensor_d0] global tensor into LDS.
// Pad: 4 DWORDs after every 32 DWORDs (=> one 16B pad per 128B row => ROWP).
// D# layout per CDNA5 ISA ch.8.  This toolchain's 6-arg builtin:
//   (u32x4 g0, i32x8 g1, i32x4 g2, i32x4 g3, i32x8 gx, i32 cpol)
// ---------------------------------------------------------------------------
static __device__ __forceinline__ void tdm_stage_tile(
    const __bf16* gsrc, unsigned lds_byte_off,
    unsigned tensor_d0, unsigned tensor_d1,
    unsigned long long d0_stride, unsigned tile_d0, unsigned tile_d1) {
#if HAVE_TDM
    const unsigned long long ga = (unsigned long long)(uintptr_t)gsrc;
    u32x4 g0;
    g0[0] = 1u;                                        // count=1, user D#
    g0[1] = lds_byte_off;                              // lds_addr
    g0[2] = (unsigned)(ga & 0xffffffffu);              // global_addr[31:0]
    g0[3] = (unsigned)((ga >> 32) & 0x01ffffffu)       // global_addr[56:32]
          | (2u << 30);                                // type = 2 ("image")
    i32x8 g1;
    g1[0] = (int)((1u << 16)    // data_size = 2 bytes
                | (1u << 20)    // pad_enable
                | (4u << 22)    // pad_interval: 32 DWORDs (one row)
                | (3u << 25));  // pad_amount  : 4 DWORDs (16B)
    g1[1] = (int)((tensor_d0 & 0xffffu) << 16);
    g1[2] = (int)(((tensor_d0 >> 16) & 0xffffu) | ((tensor_d1 & 0xffffu) << 16));
    g1[3] = (int)(((tensor_d1 >> 16) & 0xffffu) | ((tile_d0 & 0xffffu) << 16));
    g1[4] = (int)(tile_d1 & 0xffffu);                  // tile_dim1 (tile_dim2=0)
    g1[5] = (int)(d0_stride & 0xffffffffu);            // dim0 stride lo
    g1[6] = (int)((d0_stride >> 32) & 0xffffu);        // dim0 stride hi
    g1[7] = 0;
    const i32x4 z4 = {0, 0, 0, 0};
    const i32x8 z8 = {0, 0, 0, 0, 0, 0, 0, 0};
    __builtin_amdgcn_tensor_load_to_lds(g0, g1, z4, z4, z8, 0);
#else
    (void)gsrc; (void)lds_byte_off; (void)tensor_d0; (void)tensor_d1;
    (void)d0_stride; (void)tile_d0; (void)tile_d1;
#endif
}

// Fallback staging: 256 threads copy one 64-elem bf16 row each into padded LDS.
static __device__ __forceinline__ void stage_tile_fallback(
    const __bf16* __restrict__ gsrc, size_t row_stride,
    __bf16* lds_tile, int tid) {
    const __bf16* src = gsrc + (size_t)tid * row_stride;
    __bf16* dst = lds_tile + tid * ROWP;
#pragma unroll
    for (int c = 0; c < KSTEP; c += 8)
        *(v8bf*)(dst + c) = *(const v8bf*)(src + c);
}

static __device__ __forceinline__ unsigned lds_offset_of(const void* p) {
    return (unsigned)(uintptr_t)(__attribute__((address_space(3))) const void*)p;
}

// ---------------------------------------------------------------------------
// Kernel 1: dinv[i] = rsqrt( 1 + sum_j adj[i,j] )   (A = adj + I)
// ---------------------------------------------------------------------------
__global__ __launch_bounds__(256) void rowsum_dinv_kernel(
    const float* __restrict__ adj, float* __restrict__ dinv) {
    __shared__ float sm[256];
    const int i = blockIdx.x;
    const int tid = threadIdx.x;
    const float* row = adj + (size_t)i * GN;
    float s = 0.0f;
#pragma unroll
    for (int k = 0; k < GN / 256; ++k) s += row[tid + k * 256];
    sm[tid] = s;
    __syncthreads();
    for (int off = 128; off > 0; off >>= 1) {
        if (tid < off) sm[tid] += sm[tid + off];
        __syncthreads();
    }
    if (tid == 0) dinv[i] = rsqrtf(sm[0] + 1.0f);
}

// ---------------------------------------------------------------------------
// Kernel 2: W (f32 [256,1024]) -> bf16 copy
// ---------------------------------------------------------------------------
__global__ __launch_bounds__(256) void convw_kernel(
    const float* __restrict__ W, __bf16* __restrict__ Wb) {
    const size_t t = (size_t)blockIdx.x * 256 + threadIdx.x;
    Wb[t] = f2bf(W[t]);
}

// ---------------------------------------------------------------------------
// Kernel 3: Gumbel top-32 per row of the zero-diag normalized Laplacian.
// ---------------------------------------------------------------------------
static __device__ __forceinline__ float hash_uniform(unsigned v) {
    v ^= v >> 16; v *= 0x7feb352dU;
    v ^= v >> 15; v *= 0x846ca68bU;
    v ^= v >> 16;
    return ((float)(v >> 8) + 0.5f) * (1.0f / 16777216.0f); // (0,1)
}

__global__ __launch_bounds__(256) void sample_topk_kernel(
    const float* __restrict__ adj, const float* __restrict__ dinv,
    int* __restrict__ idx_out, float* __restrict__ val_out) {
    __shared__ float s_val[256];
    __shared__ int   s_idx[256];

    const int i = blockIdx.x;
    const int tid = threadIdx.x;
    const float di = dinv[i];
    const float* row = adj + (size_t)i * GN;

    float logit[GN / 256];
#pragma unroll
    for (int s = 0; s < GN / 256; ++s) {
        const int j = tid + s * 256;
        float w = (j == i) ? 0.0f : di * row[j] * dinv[j];
        unsigned h = (unsigned)(i * GN + j) * 0x9e3779b1U + 42U * 0x85ebca77U;
        const float u = hash_uniform(h);
        const float g = -__logf(-__logf(u));
        logit[s] = (w > 0.0f) ? (__logf(w) + g) : -INFINITY;
    }

    unsigned excl = 0u;
    for (int sel = 0; sel < GK; ++sel) {
        float best = -INFINITY;
        int bestj = tid;
#pragma unroll
        for (int s = 0; s < GN / 256; ++s) {
            if (!(excl & (1u << s)) && logit[s] > best) {
                best = logit[s];
                bestj = tid + s * 256;
            }
        }
        s_val[tid] = best;
        s_idx[tid] = bestj;
        __syncthreads();
        for (int off = 128; off > 0; off >>= 1) {
            if (tid < off && s_val[tid + off] > s_val[tid]) {
                s_val[tid] = s_val[tid + off];
                s_idx[tid] = s_idx[tid + off];
            }
            __syncthreads();
        }
        const int jw = s_idx[0];
        if (tid == 0) {
            idx_out[i * GK + sel] = jw;
            const float wv = (jw == i) ? 0.0f : di * row[jw] * dinv[jw];
            val_out[i * GK + sel] = (wv > 0.0f) ? wv : 0.0f;
        }
        if ((jw & 255) == tid) excl |= 1u << (jw >> 8);
        __syncthreads();
    }
}

// ---------------------------------------------------------------------------
// Kernel 4: Chebyshev step.  t_next = alpha*(ls @ t_prev) - beta*t_sub.
// Packs xc[i, f*4 + korder] = bf16(theta[korder] * t_next)  (+ theta0*x once).
// ---------------------------------------------------------------------------
__global__ __launch_bounds__(256) void cheb_prop_kernel(
    const float* __restrict__ vals, const int* __restrict__ idx,
    const float* __restrict__ tprev, const float* __restrict__ tsub,
    float alpha, float beta,
    const float* __restrict__ x0, const float* __restrict__ thetas, int korder,
    float* __restrict__ tout, __bf16* __restrict__ xc) {
    __shared__ float sv[GK];
    __shared__ int   si[GK];
    const int i = blockIdx.x;
    const int f = threadIdx.x;
    if (f < GK) { sv[f] = vals[i * GK + f]; si[f] = idx[i * GK + f]; }
    __syncthreads();

    float acc = 0.0f;
#pragma unroll
    for (int s = 0; s < GK; ++s)
        acc += sv[s] * tprev[(size_t)si[s] * GF + f];

    const float tn = alpha * acc - beta * tsub[(size_t)i * GF + f];
    tout[(size_t)i * GF + f] = tn;
    xc[(size_t)i * GKX + f * GORDER + korder] = f2bf(thetas[korder] * tn);
    if (x0)
        xc[(size_t)i * GKX + f * GORDER] = f2bf(thetas[0] * x0[(size_t)i * GF + f]);
}

// ---------------------------------------------------------------------------
// Kernel 5: yT[n,m] = bf16( sum_k xc[m,k]*W[n,k] + b[n] )
// Block = 8 waves -> 32M x 256N tile; wave = 16M x 64N; K staged 64 at a time.
// ---------------------------------------------------------------------------
__global__ __launch_bounds__(256) void gemm_xw_kernel(
    const __bf16* __restrict__ xc, const __bf16* __restrict__ Wb,
    const float* __restrict__ bias, __bf16* __restrict__ yT) {
    __shared__ __bf16 Bs[2][GF][ROWP];   // 2 x 36KB

    const int tid    = threadIdx.x;
    const int m0     = blockIdx.x * 32;
    const int lane   = tid & 31;
    const int wave   = tid >> 5;
    const int laneM  = lane & 15;
    const int laneHi = lane >> 4;
    const int waveM0 = m0 + (wave >> 2) * 16;
    const int waveN0 = (wave & 3) * 64;

    v8f acc[4] = {};
    const __bf16* arow = xc + (size_t)(waveM0 + laneM) * GKX + laneHi * 8;

    // stage first W K-tile + preload first A fragments
#if HAVE_TDM
    if (wave == 0)
        tdm_stage_tile(Wb, lds_offset_of(&Bs[0][0][0]), GKX, GF, GKX, KSTEP, GF);
    if (wave == 0) __builtin_amdgcn_s_wait_tensorcnt(0);
#else
    stage_tile_fallback(Wb, GKX, &Bs[0][0][0], tid);
#endif
    v16bf aPre0 = ld_frag(arow);
    v16bf aPre1 = ld_frag(arow + 32);
    __syncthreads();

    int cur = 0;
    for (int kk = 0; kk < GKX; kk += KSTEP) {
        const bool more = (kk + KSTEP) < GKX;
        if (more) {
#if HAVE_TDM
            if (wave == 0)
                tdm_stage_tile(Wb + kk + KSTEP, lds_offset_of(&Bs[cur ^ 1][0][0]),
                               GKX, GF, GKX, KSTEP, GF);
#else
            stage_tile_fallback(Wb + kk + KSTEP, GKX, &Bs[cur ^ 1][0][0], tid);
#endif
        }

        // software-pipelined A fragments
        const v16bf a0 = aPre0;
        const v16bf a1 = aPre1;
        if (more) {
            aPre0 = ld_frag(arow + kk + KSTEP);
            aPre1 = ld_frag(arow + kk + KSTEP + 32);
        }

#pragma unroll
        for (int sub = 0; sub < 2; ++sub) {
            const v16bf a = sub ? a1 : a0;
            v16bf b[4];
#pragma unroll
            for (int t = 0; t < 4; ++t)
                b[t] = ld_frag(&Bs[cur][waveN0 + t * 16 + laneM][sub * 32 + laneHi * 8]);
#pragma unroll
            for (int t = 0; t < 4; ++t)
                acc[t] = __builtin_amdgcn_wmma_f32_16x16x32_bf16(
                    false, a, false, b[t], (short)0, acc[t], false, false);
        }

#if HAVE_TDM
        if (wave == 0) __builtin_amdgcn_s_wait_tensorcnt(0);
#endif
        __syncthreads();
        cur ^= 1;
    }

#pragma unroll
    for (int t = 0; t < 4; ++t) {
        const int n = waveN0 + t * 16 + laneM;
        const float bv = bias[n];
#pragma unroll
        for (int r = 0; r < 8; ++r) {
            const int m = waveM0 + r + laneHi * 8;
            yT[(size_t)n * GN + m] = f2bf(acc[t][r] + bv);
        }
    }
}

// ---------------------------------------------------------------------------
// Kernel 6: out[m,n] = dinv[m] * sum_j ( adj[m,j]*dinv[j]*(j!=m) ) * y[j,n]
// adj streamed f32 (software-pipelined, prefetched), converted to bf16
// Laplacian fragments in registers; yT K-tiles TDM-staged in LDS; dinv in LDS.
// ---------------------------------------------------------------------------
struct RawA { float4 a0, a1, a2, a3; };

static __device__ __forceinline__ RawA load_rawA(const float* __restrict__ arow,
                                                 int c0) {
    RawA r;
    r.a0 = *(const float4*)(arow + c0);
    r.a1 = *(const float4*)(arow + c0 + 4);
    r.a2 = *(const float4*)(arow + c0 + 16);
    r.a3 = *(const float4*)(arow + c0 + 20);
    return r;
}

// Build a 16x32 Laplacian A-fragment: a[e] = bf16(adj*dinv_j), diag zeroed.
static __device__ __forceinline__ v16bf make_lap_frag(
    const RawA& raw, const float* Dv, int c0, int rowM) {
    const float4 d0 = *(const float4*)(Dv + c0);
    const float4 d1 = *(const float4*)(Dv + c0 + 4);
    const float4 d2 = *(const float4*)(Dv + c0 + 16);
    const float4 d3 = *(const float4*)(Dv + c0 + 20);
    const float af[16] = { raw.a0.x, raw.a0.y, raw.a0.z, raw.a0.w,
                           raw.a1.x, raw.a1.y, raw.a1.z, raw.a1.w,
                           raw.a2.x, raw.a2.y, raw.a2.z, raw.a2.w,
                           raw.a3.x, raw.a3.y, raw.a3.z, raw.a3.w };
    const float df[16] = { d0.x, d0.y, d0.z, d0.w, d1.x, d1.y, d1.z, d1.w,
                           d2.x, d2.y, d2.z, d2.w, d3.x, d3.y, d3.z, d3.w };
    v16bf a;
#pragma unroll
    for (int e = 0; e < 16; ++e) {
        const int j = c0 + ((e < 8) ? e : e + 8);
        a[e] = f2bf((j == rowM) ? 0.0f : af[e] * df[e]);
    }
    return a;
}

__global__ __launch_bounds__(256) void gemm_ly_kernel(
    const float* __restrict__ adj, const float* __restrict__ dinv,
    const __bf16* __restrict__ yT, float* __restrict__ out) {
    __shared__ __bf16 Bs[2][GF][ROWP];   // 2 x 36KB
    __shared__ float  Dv[GN];            // 24KB

    const int tid    = threadIdx.x;
    const int m0     = blockIdx.x * 32;
    const int lane   = tid & 31;
    const int wave   = tid >> 5;
    const int laneM  = lane & 15;
    const int laneHi = lane >> 4;
    const int waveM0 = m0 + (wave >> 2) * 16;
    const int waveN0 = (wave & 3) * 64;

    // stage dinv once (L2-resident broadcast -> LDS)
#pragma unroll
    for (int k = 0; k < GN / 256; ++k) Dv[tid + k * 256] = dinv[tid + k * 256];

    const int rowM = waveM0 + laneM;
    const float* arow = adj + (size_t)rowM * GN;

    v8f acc[4] = {};

    // stage first yT K-tile + preload first adj fragments
#if HAVE_TDM
    if (wave == 0)
        tdm_stage_tile(yT, lds_offset_of(&Bs[0][0][0]), GN, GF, GN, KSTEP, GF);
    if (wave == 0) __builtin_amdgcn_s_wait_tensorcnt(0);
#else
    stage_tile_fallback(yT, GN, &Bs[0][0][0], tid);
#endif
    RawA pre0 = load_rawA(arow, laneHi * 8);
    RawA pre1 = load_rawA(arow, 32 + laneHi * 8);
    __syncthreads();

    int cur = 0;
    for (int kk = 0; kk < GN; kk += KSTEP) {
        const bool more = (kk + KSTEP) < GN;
        const int c0 = kk + laneHi * 8;

        if (more) {
#if HAVE_TDM
            if (wave == 0)
                tdm_stage_tile(yT + kk + KSTEP, lds_offset_of(&Bs[cur ^ 1][0][0]),
                               GN, GF, GN, KSTEP, GF);
#else
            stage_tile_fallback(yT + kk + KSTEP, GN, &Bs[cur ^ 1][0][0], tid);
#endif
        }

        // software pipeline the streamed adj row
        const RawA raw0 = pre0;
        const RawA raw1 = pre1;
        if (more) {
            __builtin_prefetch(arow + c0 + 2 * KSTEP, 0, 0);
            pre0 = load_rawA(arow, c0 + KSTEP);
            pre1 = load_rawA(arow, c0 + KSTEP + 32);
        }

#pragma unroll
        for (int sub = 0; sub < 2; ++sub) {
            const v16bf a = make_lap_frag(sub ? raw1 : raw0, Dv,
                                          c0 + sub * 32, rowM);
            v16bf b[4];
#pragma unroll
            for (int t = 0; t < 4; ++t)
                b[t] = ld_frag(&Bs[cur][waveN0 + t * 16 + laneM][sub * 32 + laneHi * 8]);
#pragma unroll
            for (int t = 0; t < 4; ++t)
                acc[t] = __builtin_amdgcn_wmma_f32_16x16x32_bf16(
                    false, a, false, b[t], (short)0, acc[t], false, false);
        }

#if HAVE_TDM
        if (wave == 0) __builtin_amdgcn_s_wait_tensorcnt(0);
#endif
        __syncthreads();
        cur ^= 1;
    }

#pragma unroll
    for (int t = 0; t < 4; ++t) {
        const int n = waveN0 + t * 16 + laneM;
#pragma unroll
        for (int r = 0; r < 8; ++r) {
            const int m = waveM0 + r + laneHi * 8;
            out[(size_t)m * GF + n] = acc[t][r] * Dv[m];
        }
    }
}

// ---------------------------------------------------------------------------
// Host-side launcher
// ---------------------------------------------------------------------------
extern "C" void kernel_launch(void* const* d_in, const int* in_sizes, int n_in,
                              void* d_out, int out_size, void* d_ws, size_t ws_size,
                              hipStream_t stream) {
    const float* x      = (const float*)d_in[0];   // [6144, 256]
    const float* adj    = (const float*)d_in[1];   // [6144, 6144]
    const float* thetas = (const float*)d_in[2];   // [4]
    const float* W      = (const float*)d_in[3];   // [256, 1024]
    const float* bias   = (const float*)d_in[4];   // [256]
    float* out          = (float*)d_out;           // [6144, 256]

    char* ws = (char*)d_ws;
    size_t off = 0;
    auto alloc = [&](size_t bytes) -> char* {
        char* p = ws + off;
        off += (bytes + 255) & ~(size_t)255;
        return p;
    };
    float*  dinv = (float*)alloc((size_t)GN * 4);
    int*    sidx = (int*)  alloc((size_t)GN * GK * 4);
    float*  sval = (float*)alloc((size_t)GN * GK * 4);
    float*  t1   = (float*)alloc((size_t)GN * GF * 4);
    float*  t2   = (float*)alloc((size_t)GN * GF * 4);
    float*  t3   = (float*)alloc((size_t)GN * GF * 4);
    __bf16* xc   = (__bf16*)alloc((size_t)GN * GKX * 2);
    __bf16* Wb   = (__bf16*)alloc((size_t)GF * GKX * 2);
    __bf16* yT   = (__bf16*)alloc((size_t)GF * GN * 2);
    (void)ws_size; (void)in_sizes; (void)n_in; (void)out_size;

    rowsum_dinv_kernel<<<GN, 256, 0, stream>>>(adj, dinv);
    convw_kernel<<<(GF * GKX) / 256, 256, 0, stream>>>(W, Wb);
    sample_topk_kernel<<<GN, 256, 0, stream>>>(adj, dinv, sidx, sval);

    cheb_prop_kernel<<<GN, 256, 0, stream>>>(sval, sidx, x, x, 1.0f, 0.0f,
                                             x, thetas, 1, t1, xc);
    cheb_prop_kernel<<<GN, 256, 0, stream>>>(sval, sidx, t1, x, 2.0f, 1.0f,
                                             nullptr, thetas, 2, t2, xc);
    cheb_prop_kernel<<<GN, 256, 0, stream>>>(sval, sidx, t2, t1, 2.0f, 1.0f,
                                             nullptr, thetas, 3, t3, xc);

    gemm_xw_kernel<<<GN / 32, 256, 0, stream>>>(xc, Wb, bias, yT);
    gemm_ly_kernel<<<GN / 32, 256, 0, stream>>>(adj, dinv, yT, out);
}
